// MultiHeadAttention_72859825209472
// MI455X (gfx1250) — matrix operand order repaired and checked by reading, not compiled, not run on hardware
//
#include <hip/hip_runtime.h>

// ---------------------------------------------------------------------------
// MI455X (gfx1250) multi-head attention, bf16 WMMA pipeline.
// Compute-bound (~412 GFLOP vs ~0.4 GB traffic) => route every GEMM through
// v_wmma_f32_16x16x32_bf16, and stage GEMM tiles with the Tensor Data Mover
// (tensor_load_to_lds + s_wait_tensorcnt), double-buffered in LDS.
// wave32 everywhere.
// ---------------------------------------------------------------------------

typedef __attribute__((ext_vector_type(16))) __bf16        v16bf;
typedef __attribute__((ext_vector_type(8)))  float         v8f;
typedef __attribute__((ext_vector_type(4)))  unsigned int  v4u;
typedef __attribute__((ext_vector_type(8)))  int           v8i;
typedef __attribute__((ext_vector_type(4)))  int           v4i;
typedef unsigned short ushort_t;

#define D_MODEL   2048
#define NUM_HEADS 16
#define D_K       128
#define SEQ       2048
#define BATCH     4
#define MROWS     (BATCH * SEQ)   // 8192

static __device__ __forceinline__ ushort_t f32_to_bf16(float f) {
  union { float f; unsigned u; } v; v.f = f;
  unsigned r = (v.u + 0x7FFFu + ((v.u >> 16) & 1u)) >> 16;  // RNE
  return (ushort_t)r;
}
static __device__ __forceinline__ float bf16_to_f32(ushort_t h) {
  union { unsigned u; float f; } v; v.u = ((unsigned)h) << 16;
  return v.f;
}

union BF16Frag { v16bf v; v4u q[2]; };

static __device__ __forceinline__ v8f wmma_bf16(v16bf a, v16bf b, v8f c) {
  // (neg_a, A, neg_b, B, c_mod, C, reuse_a, reuse_b)
  return __builtin_amdgcn_wmma_f32_16x16x32_bf16(false, a, false, b,
                                                 (short)0, c, false, false);
}

// ---------------------------------------------------------------------------
// TDM: DMA a 2D bf16 tile [128 rows x 32 cols] from a row-major matrix with
// row stride D_MODEL into LDS (rows packed contiguously: lds[r*32 + c]).
// D# encoding per CDNA5 ISA ch.8 (group0: count/lds/global/type, group1:
// data_size=2B, tensor dims, tile dims, dim0 stride). Tracked by TENSORcnt.
// ---------------------------------------------------------------------------
static __device__ __forceinline__ void tdm_load_tile(
    unsigned lds_byte_off, const ushort_t* tile_origin, unsigned rows_total) {
  unsigned long long ga = (unsigned long long)(size_t)tile_origin;
  v4u g0;
  g0.x = 1u;                                   // count=1, is_restore=0
  g0.y = lds_byte_off;                         // [63:32]  lds_addr
  g0.z = (unsigned)ga;                         // [95:64]  global_addr lo
  g0.w = (unsigned)(ga >> 32) | 0x80000000u;   // [120:96] addr hi | type=2

  v8i g1;
  g1[0] = 0x00010000;                          // mask=0, data_size=1 (2 bytes)
  g1[1] = (int)(((unsigned)D_MODEL & 0xFFFFu) << 16);        // dim0 lo16
  g1[2] = (int)(((unsigned)D_MODEL >> 16) |
                ((rows_total & 0xFFFFu) << 16));             // dim0 hi / dim1 lo
  g1[3] = (int)((rows_total >> 16) | (32u << 16));           // dim1 hi / tile0=32
  g1[4] = 128;                                 // tile_dim1=128, tile_dim2=0
  g1[5] = D_MODEL;                             // tensor_dim0_stride lo
  g1[6] = 0;                                   // stride hi / dim1_stride lo
  g1[7] = 0;

  v4i gz = {0, 0, 0, 0};
#if __clang_major__ >= 23
  v8i gz8 = {0, 0, 0, 0, 0, 0, 0, 0};
  __builtin_amdgcn_tensor_load_to_lds(g0, g1, gz, gz, gz8, 0);
#else
  __builtin_amdgcn_tensor_load_to_lds(g0, g1, gz, gz, 0);
#endif
}

// ---------------------------------------------------------------------------
// fp32 -> bf16 conversion
// ---------------------------------------------------------------------------
__global__ void cvt_f32_bf16(const float* __restrict__ in,
                             ushort_t* __restrict__ out, int n) {
  int i = blockIdx.x * blockDim.x + threadIdx.x;
  if (i < n) out[i] = f32_to_bf16(in[i]);
}

// ---------------------------------------------------------------------------
// Tiled GEMM: out[m,n] = sum_k A[m,k] * W[n,k] + bias[n]
// A: [8192,2048] bf16, W: [2048,2048] bf16 (row-major => implicit W^T GEMM).
// Block 128x128, BK=32, 8 waves in 4(M) x 2(N); each wave: 2x4 WMMA tiles.
// Global->LDS staging via TDM tensor_load_to_lds, double-buffered:
// wave0 DMAs the A tile, wave1 the B tile, prefetching one BK step ahead.
// mode 0: bf16 row-major out; mode 1: bf16 scatter to Vt[b,h,d,s];
// mode 2: fp32 row-major out.
// ---------------------------------------------------------------------------
__global__ __launch_bounds__(256) void gemm_bf16(
    const ushort_t* __restrict__ A, const ushort_t* __restrict__ W,
    ushort_t* __restrict__ outb, float* __restrict__ outf,
    const float* __restrict__ bias, int mode) {
  constexpr int Kd = D_MODEL, Nd = D_MODEL;
  __shared__ __align__(16) ushort_t As[2][128 * 32];
  __shared__ __align__(16) ushort_t Bs[2][128 * 32];

  const int tid = threadIdx.x;
  const int wid = tid >> 5, lane = tid & 31;
  const int lrow = lane & 15, lhi = lane >> 4;
  const int waveM = wid & 3, waveN = wid >> 2;
  const int row0 = blockIdx.y * 128, col0 = blockIdx.x * 128;

  const unsigned aoff0 = (unsigned)(size_t)&As[0][0];
  const unsigned aoff1 = (unsigned)(size_t)&As[1][0];
  const unsigned boff0 = (unsigned)(size_t)&Bs[0][0];
  const unsigned boff1 = (unsigned)(size_t)&Bs[1][0];

  v8f acc[2][4];
#pragma unroll
  for (int i = 0; i < 2; ++i)
#pragma unroll
    for (int j = 0; j < 4; ++j) acc[i][j] = {};

  // prologue: DMA tile k0=0 into buffer 0
  if (wid == 0) {
    tdm_load_tile(aoff0, A + (size_t)row0 * Kd, MROWS);
  } else if (wid == 1) {
    tdm_load_tile(boff0, W + (size_t)col0 * Kd, D_MODEL);
  }

  int buf = 0;
  for (int k0 = 0; k0 < Kd; k0 += 32) {
    // prefetch next tile into the other buffer, then retire current tile
    if (wid == 0) {
      if (k0 + 32 < Kd) {
        tdm_load_tile(buf ? aoff0 : aoff1, A + (size_t)row0 * Kd + k0 + 32,
                      MROWS);
        __builtin_amdgcn_s_wait_tensorcnt(1);
      } else {
        __builtin_amdgcn_s_wait_tensorcnt(0);
      }
    } else if (wid == 1) {
      if (k0 + 32 < Kd) {
        tdm_load_tile(buf ? boff0 : boff1, W + (size_t)col0 * Kd + k0 + 32,
                      D_MODEL);
        __builtin_amdgcn_s_wait_tensorcnt(1);
      } else {
        __builtin_amdgcn_s_wait_tensorcnt(0);
      }
    }
    __syncthreads();  // publish DMA'd tile to all waves

    const ushort_t* Asb = &As[buf][0];
    const ushort_t* Bsb = &Bs[buf][0];
    BF16Frag af[2], bfr[4];
#pragma unroll
    for (int i = 0; i < 2; ++i) {  // A frag: row per lane, 16 contiguous K
      int r = waveM * 32 + i * 16 + lrow;
      af[i].q[0] = *(const v4u*)&Asb[r * 32 + 16 * lhi];
      af[i].q[1] = *(const v4u*)&Asb[r * 32 + 16 * lhi + 8];
    }
#pragma unroll
    for (int j = 0; j < 4; ++j) {  // B frag: col per lane, 16 contiguous K
      int c = waveN * 64 + j * 16 + lrow;
      bfr[j].q[0] = *(const v4u*)&Bsb[c * 32 + 16 * lhi];
      bfr[j].q[1] = *(const v4u*)&Bsb[c * 32 + 16 * lhi + 8];
    }
#pragma unroll
    for (int i = 0; i < 2; ++i)
#pragma unroll
      for (int j = 0; j < 4; ++j)
        acc[i][j] = wmma_bf16(af[i].v, bfr[j].v, acc[i][j]);
    __syncthreads();  // all reads done before this buffer is DMA'd again
    buf ^= 1;
  }

  // epilogue: C-layout row = e + 8*lhi, col = lrow
#pragma unroll
  for (int i = 0; i < 2; ++i) {
#pragma unroll
    for (int j = 0; j < 4; ++j) {
#pragma unroll
      for (int e = 0; e < 8; ++e) {
        int r = row0 + waveM * 32 + i * 16 + e + 8 * lhi;
        int c = col0 + waveN * 64 + j * 16 + lrow;
        float val = acc[i][j][e] + bias[c];
        if (mode == 0) {
          outb[(size_t)r * Nd + c] = f32_to_bf16(val);
        } else if (mode == 1) {  // scatter to Vt[b,h,d,s]
          int bb = r >> 11, s = r & (SEQ - 1);
          int h = c >> 7, d = c & (D_K - 1);
          outb[((size_t)(bb * NUM_HEADS + h) * D_K + d) * SEQ + s] =
              f32_to_bf16(val);
        } else {
          outf[(size_t)r * Nd + c] = val;
        }
      }
    }
  }
}

// ---------------------------------------------------------------------------
// Rotate-half RoPE in place on bf16 [B,S,H,d_k] (== [B,S,D] row-major).
// ---------------------------------------------------------------------------
__global__ void rope_bf16(ushort_t* __restrict__ t, int ntot) {
  int i = blockIdx.x * blockDim.x + threadIdx.x;
  if (i >= ntot) return;
  int d = i & 63;                        // pair index, d_k/2 = 64
  int h = (i >> 6) & (NUM_HEADS - 1);
  int s = (i >> 10) & (SEQ - 1);
  int b = i >> 21;
  size_t base = ((size_t)(b * SEQ + s)) * D_MODEL + h * D_K;
  // inv_freq = 10000^(-d/64) = 2^(-d * log2(10000)/64)
  float freq = exp2f(-(float)d * (13.287712379549449f / 64.f));
  float ang = (float)s * freq;
  float c = cosf(ang), sn = sinf(ang);
  float x1 = bf16_to_f32(t[base + d]);
  float x2 = bf16_to_f32(t[base + d + 64]);
  t[base + d]      = f32_to_bf16(x1 * c - x2 * sn);
  t[base + d + 64] = f32_to_bf16(x2 * c + x1 * sn);
}

// ---------------------------------------------------------------------------
// Flash attention: one block = (q-tile of 128) x (b,h); 8 waves x 16 q-rows.
// Streams kpos in blocks of 32: QK^T (8 WMMAs) -> online softmax (shfl_xor
// row reductions across 16-lane halves) -> P restaged via wave-private LDS
// into A-fragment layout -> PV (8 WMMAs into 16x128 accumulator).
// ---------------------------------------------------------------------------
__global__ __launch_bounds__(256) void flash_attn(
    const ushort_t* __restrict__ Q, const ushort_t* __restrict__ Kk,
    const ushort_t* __restrict__ Vt, ushort_t* __restrict__ O) {
  __shared__ __align__(16) ushort_t psmem[8][16 * 32];

  const int tid = threadIdx.x, wid = tid >> 5, lane = tid & 31;
  const int lrow = lane & 15, lhi = lane >> 4;
  const int bh = blockIdx.y;
  const int b = bh >> 4, h = bh & (NUM_HEADS - 1);
  const int q0 = blockIdx.x * 128 + wid * 16;

  const ushort_t* Qb = Q + (size_t)b * SEQ * D_MODEL + h * D_K;
  const ushort_t* Kb = Kk + (size_t)b * SEQ * D_MODEL + h * D_K;
  const ushort_t* Vb = Vt + (size_t)bh * D_K * SEQ;

  // Preload Q A-fragments (loop invariant): 4 chunks of 32 along d
  BF16Frag qa[4];
#pragma unroll
  for (int t = 0; t < 4; ++t) {
    const ushort_t* p = Qb + (size_t)(q0 + lrow) * D_MODEL + t * 32 + 16 * lhi;
    qa[t].q[0] = *(const v4u*)p;
    qa[t].q[1] = *(const v4u*)(p + 8);
  }

  v8f o[8];
#pragma unroll
  for (int f = 0; f < 8; ++f) o[f] = {};
  float mrun[8], lrun[8];
#pragma unroll
  for (int j = 0; j < 8; ++j) { mrun[j] = -INFINITY; lrun[j] = 0.f; }

  const float scale = 0.08838834764831845f;  // 1/sqrt(128)

  for (int kp = 0; kp < SEQ; kp += 32) {
    // ---- scores for kpos columns [kp, kp+32) ----
    v8f s0 = {}, s1 = {};
#pragma unroll
    for (int t = 0; t < 4; ++t) {
      BF16Frag kb0, kb1;
      const ushort_t* p0 = Kb + (size_t)(kp + lrow) * D_MODEL + t * 32 + 16 * lhi;
      kb0.q[0] = *(const v4u*)p0;
      kb0.q[1] = *(const v4u*)(p0 + 8);
      const ushort_t* p1 = p0 + (size_t)16 * D_MODEL;
      kb1.q[0] = *(const v4u*)p1;
      kb1.q[1] = *(const v4u*)(p1 + 8);
      s0 = wmma_bf16(qa[t].v, kb0.v, s0);
      s1 = wmma_bf16(qa[t].v, kb1.v, s1);
    }

    // ---- online softmax (rows spread across the 16 lanes of a half) ----
    float corr[8];
#pragma unroll
    for (int j = 0; j < 8; ++j) {
      float a = s0[j] * scale, bb = s1[j] * scale;
      float tmax = fmaxf(a, bb);
#pragma unroll
      for (int msk = 1; msk < 16; msk <<= 1)
        tmax = fmaxf(tmax, __shfl_xor(tmax, msk, 32));
      float mnew = fmaxf(mrun[j], tmax);
      float p0v = __expf(a - mnew);
      float p1v = __expf(bb - mnew);
      float rs = p0v + p1v;
#pragma unroll
      for (int msk = 1; msk < 16; msk <<= 1) rs += __shfl_xor(rs, msk, 32);
      corr[j] = __expf(mrun[j] - mnew);
      lrun[j] = lrun[j] * corr[j] + rs;
      mrun[j] = mnew;
      int prow = (j + 8 * lhi) * 32 + lrow;  // C-layout -> [row][kcol]
      psmem[wid][prow]      = f32_to_bf16(p0v);
      psmem[wid][prow + 16] = f32_to_bf16(p1v);
    }

    // ---- restage P as A-fragment (wave-private LDS; DS ops in-order) ----
    BF16Frag pa;
    pa.q[0] = *(const v4u*)&psmem[wid][lrow * 32 + 16 * lhi];
    pa.q[1] = *(const v4u*)&psmem[wid][lrow * 32 + 16 * lhi + 8];

    // ---- o = o*corr + P @ V ----
#pragma unroll
    for (int f = 0; f < 8; ++f) {
#pragma unroll
      for (int j = 0; j < 8; ++j) o[f][j] *= corr[j];
      BF16Frag vb;
      const ushort_t* pv = Vb + (size_t)(f * 16 + lrow) * SEQ + kp + 16 * lhi;
      vb.q[0] = *(const v4u*)pv;
      vb.q[1] = *(const v4u*)(pv + 8);
      o[f] = wmma_bf16(pa.v, vb.v, o[f]);
    }
  }

#pragma unroll
  for (int j = 0; j < 8; ++j) lrun[j] = 1.f / lrun[j];
#pragma unroll
  for (int f = 0; f < 8; ++f) {
#pragma unroll
    for (int j = 0; j < 8; ++j) {
      int r = q0 + j + 8 * lhi;
      int c = h * D_K + f * 16 + lrow;
      O[(size_t)(b * SEQ + r) * D_MODEL + c] = f32_to_bf16(o[f][j] * lrun[j]);
    }
  }
}

// ---------------------------------------------------------------------------
// Launcher
// ---------------------------------------------------------------------------
extern "C" void kernel_launch(void* const* d_in, const int* in_sizes, int n_in,
                              void* d_out, int out_size, void* d_ws,
                              size_t ws_size, hipStream_t stream) {
  const float* x  = (const float*)d_in[0];
  const float* Wq = (const float*)d_in[1];
  const float* bq = (const float*)d_in[2];
  const float* Wk = (const float*)d_in[3];
  const float* bk = (const float*)d_in[4];
  const float* Wv = (const float*)d_in[5];
  const float* bv = (const float*)d_in[6];
  const float* Wo = (const float*)d_in[7];
  const float* bo = (const float*)d_in[8];
  float* out = (float*)d_out;

  char* ws = (char*)d_ws;
  ushort_t* xb  = (ushort_t*)(ws + 0);           // 32 MiB
  ushort_t* wqb = (ushort_t*)(ws + 33554432);    // 8 MiB each
  ushort_t* wkb = (ushort_t*)(ws + 41943040);
  ushort_t* wvb = (ushort_t*)(ws + 50331648);
  ushort_t* wob = (ushort_t*)(ws + 58720256);
  ushort_t* Qb  = (ushort_t*)(ws + 67108864);    // 32 MiB each
  ushort_t* Kbf = (ushort_t*)(ws + 100663296);
  ushort_t* Vtb = (ushort_t*)(ws + 134217728);
  ushort_t* Ab  = (ushort_t*)(ws + 167772160);   // end ~192 MiB

  const int nx = MROWS * D_MODEL;      // 16,777,216
  const int nw = D_MODEL * D_MODEL;    // 4,194,304
  cvt_f32_bf16<<<nx / 256, 256, 0, stream>>>(x, xb, nx);
  cvt_f32_bf16<<<nw / 256, 256, 0, stream>>>(Wq, wqb, nw);
  cvt_f32_bf16<<<nw / 256, 256, 0, stream>>>(Wk, wkb, nw);
  cvt_f32_bf16<<<nw / 256, 256, 0, stream>>>(Wv, wvb, nw);
  cvt_f32_bf16<<<nw / 256, 256, 0, stream>>>(Wo, wob, nw);

  dim3 ggrid(D_MODEL / 128, MROWS / 128);  // (16, 64)
  gemm_bf16<<<ggrid, 256, 0, stream>>>(xb, wqb, Qb, nullptr, bq, 0);
  gemm_bf16<<<ggrid, 256, 0, stream>>>(xb, wkb, Kbf, nullptr, bk, 0);
  gemm_bf16<<<ggrid, 256, 0, stream>>>(xb, wvb, Vtb, nullptr, bv, 1);

  const int nr = BATCH * SEQ * NUM_HEADS * 64;  // 8,388,608 rotation pairs
  rope_bf16<<<nr / 256, 256, 0, stream>>>(Qb, nr);
  rope_bf16<<<nr / 256, 256, 0, stream>>>(Kbf, nr);

  dim3 fgrid(SEQ / 128, BATCH * NUM_HEADS);  // (16, 64)
  flash_attn<<<fgrid, 256, 0, stream>>>(Qb, Kbf, Vtb, Ab);

  gemm_bf16<<<ggrid, 256, 0, stream>>>(Ab, wob, nullptr, out, bo, 2);
}
